// Alpamayo1ActionExpertPatch_12360915878716
// MI455X (gfx1250) — matrix operand order repaired and checked by reading, not compiled, not run on hardware
//
#include <hip/hip_runtime.h>
#include <hip/hip_bf16.h>

// ---------------- config (matches reference) ----------------
#define L_     2
#define D_     2048
#define HQ_    16
#define HKV_   8
#define DH_    128
#define F_     6144
#define T_     64
#define A_     8
#define PAST_  2048
#define CACHE_ (PAST_ + T_)
#define MAXPOS_ 4096
#define TF_    256

typedef __attribute__((ext_vector_type(16))) __bf16 v16bf;
typedef __attribute__((ext_vector_type(8)))  float  v8f;

// WMMA fragment index helpers (wave32; A/B use this file's contiguous k = 16*(L/16)+element, not AMD's table order):
//  A 16x32 bf16: lane L holds row M=L%16; kb = (L<16)?0:8; VGPR v half h -> K = (kb?16:0)+2v+h
//  B 32x16 bf16: lane L holds col N=L%16; kb = (L<16)?0:16; VGPR v half h -> K = kb+2v+h
//  C/D 16x16 f32: VGPR v -> (M = v + 8*(L>=16), N = L%16)
__device__ __forceinline__ int a_koff(int v, int half, int kb) {
    return (kb ? 16 : 0) + 2 * v + half;
}

// ---------------- small helpers ----------------
__global__ void copy4_kernel(const float4* __restrict__ src, float4* __restrict__ dst, int n4) {
    int i = blockIdx.x * 256 + threadIdx.x;
    if (i < n4) dst[i] = src[i];
}

// th = silu(tfeat @ Wt1), tfeat = [sin(t0*f), cos(t0*f)], f_i = 10000^{-i/128}
__global__ __launch_bounds__(256) void time_mlp1_kernel(const float* __restrict__ t0,
                                                        const float* __restrict__ Wt1,
                                                        float* __restrict__ th) {
    __shared__ float tf[TF_];
    int tid = threadIdx.x;
    {
        int i = tid & 127;
        float fr = __expf(-__logf(10000.f) * (float)i / 128.f);
        float v = t0[0] * fr;
        tf[tid] = (tid < 128) ? __sinf(v) : __cosf(v);
    }
    __syncthreads();
    int d = blockIdx.x * 256 + tid;
    float s = 0.f;
    #pragma unroll 4
    for (int i = 0; i < TF_; ++i) s += tf[i] * Wt1[(size_t)i * D_ + d];
    th[d] = s / (1.f + __expf(-s));
}

// temb = th @ Wt2
__global__ __launch_bounds__(256) void time_mlp2_kernel(const float* __restrict__ th,
                                                        const float* __restrict__ Wt2,
                                                        float* __restrict__ temb) {
    __shared__ float sh[D_];
    int tid = threadIdx.x;
    for (int i = tid; i < D_; i += 256) sh[i] = th[i];
    __syncthreads();
    int d = blockIdx.x * 256 + tid;
    float s = 0.f;
    #pragma unroll 4
    for (int i = 0; i < D_; ++i) s += sh[i] * Wt2[(size_t)i * D_ + d];
    temb[d] = s;
}

// h[t][d] = noise[t]@W_in[:,d] + b_in[d] + temb[d]
__global__ void h_init_kernel(const float* __restrict__ noise, const float* __restrict__ W_in,
                              const float* __restrict__ b_in, const float* __restrict__ temb,
                              float* __restrict__ h) {
    int d = blockIdx.x * 256 + threadIdx.x;
    int t = blockIdx.y;
    float s = b_in[d] + temb[d];
    #pragma unroll
    for (int a = 0; a < A_; ++a) s += noise[t * A_ + a] * W_in[(size_t)a * D_ + d];
    h[(size_t)t * D_ + d] = s;
}

// y = rms(x) * w   (rows of length D)
__global__ __launch_bounds__(256) void rmsnorm_kernel(const float* __restrict__ x,
                                                      const float* __restrict__ w,
                                                      float* __restrict__ y) {
    int t = blockIdx.x, tid = threadIdx.x;
    __shared__ float red[8];
    float ss = 0.f;
    for (int d = tid; d < D_; d += 256) { float v = x[(size_t)t * D_ + d]; ss += v * v; }
    for (int off = 16; off; off >>= 1) ss += __shfl_xor(ss, off);
    if ((tid & 31) == 0) red[tid >> 5] = ss;
    __syncthreads();
    float tot = 0.f;
    #pragma unroll
    for (int wv = 0; wv < 8; ++wv) tot += red[wv];
    float inv = rsqrtf(tot / (float)D_ + 1e-6f);
    for (int d = tid; d < D_; d += 256)
        y[(size_t)t * D_ + d] = x[(size_t)t * D_ + d] * inv * w[d];
}

// ---------------- the workhorse: M=64 WMMA GEMM, bf16 compute / f32 accumulate ----------------
// Out[64,N] (=|+=) A[64,K] @ W[K,N].  Block = 8 waves; each wave owns a 16-col N tile
// across all four 16-row M tiles; weights are streamed exactly once grid-wide
// (this kernel family is HBM-bound: ~400MB of fp32 weights vs a ~20us roofline floor).
template <bool ACCUM>
__global__ __launch_bounds__(256) void gemm64_wmma(const float* __restrict__ Am,
                                                   const float* __restrict__ W,
                                                   float* __restrict__ Out,
                                                   int Kdim, int Ndim) {
    __shared__ __bf16 As[64][66];   // bf16 staging, +2 pad -> 33-bank row stride
    int tid = threadIdx.x, wave = tid >> 5, lane = tid & 31;
    int ln = lane & 15, hi = lane >> 4;
    int n0 = blockIdx.x * 128 + wave * 16;
    int n = n0 + ln;

    v8f acc[4];
    #pragma unroll
    for (int mt = 0; mt < 4; ++mt) { v8f z = {}; acc[mt] = z; }

    for (int kk = 0; kk < Kdim; kk += 64) {
        __syncthreads();
        // stage A chunk [64 x 64] f32 -> bf16 LDS
        for (int i = tid; i < 64 * 16; i += 256) {
            int row = i >> 4, c4 = (i & 15) * 4;
            float4 f = *(const float4*)(Am + (size_t)row * Kdim + kk + c4);
            As[row][c4 + 0] = (__bf16)f.x;
            As[row][c4 + 1] = (__bf16)f.y;
            As[row][c4 + 2] = (__bf16)f.z;
            As[row][c4 + 3] = (__bf16)f.w;
        }
        __syncthreads();
        if (kk + 64 < Kdim)  // keep the weight stream ahead of the MACs
            __builtin_prefetch(&W[(size_t)(kk + 64 + (hi ? 16 : 0)) * Ndim + n], 0, 1);
        #pragma unroll
        for (int ks = 0; ks < 64; ks += 32) {
            v16bf bW;
            int kbb = hi ? 16 : 0;
            #pragma unroll
            for (int v = 0; v < 8; ++v) {
                size_t kr = (size_t)(kk + ks + kbb + 2 * v);
                bW[2 * v + 0] = (__bf16)W[kr * Ndim + n];
                bW[2 * v + 1] = (__bf16)W[(kr + 1) * Ndim + n];
            }
            int kb = hi ? 8 : 0;
            #pragma unroll
            for (int mt = 0; mt < 4; ++mt) {  // 4 independent accumulators: no WMMA RAW stalls
                v16bf aF;
                int r = mt * 16 + ln;
                #pragma unroll
                for (int v = 0; v < 8; ++v) {
                    int ko = a_koff(v, 0, kb) + ks;
                    aF[2 * v + 0] = As[r][ko];
                    aF[2 * v + 1] = As[r][ko + 1];
                }
                acc[mt] = __builtin_amdgcn_wmma_f32_16x16x32_bf16(
                    false, aF, false, bW, (short)0, acc[mt], false, false);
            }
        }
    }
    #pragma unroll
    for (int mt = 0; mt < 4; ++mt)
        #pragma unroll
        for (int v = 0; v < 8; ++v) {
            int row = mt * 16 + v + 8 * hi;
            size_t idx = (size_t)row * Ndim + n;
            if (ACCUM) Out[idx] += acc[mt][v];
            else       Out[idx] = acc[mt][v];
        }
}

// ---------------- q/k per-head RMSNorm + RoPE + cache write ----------------
// grid (T, HQ + 2*HKV), block 128 (one head vector)
__global__ __launch_bounds__(128) void qk_post_kernel(float* __restrict__ qb,
                                                      const float* __restrict__ kraw,
                                                      const float* __restrict__ vraw,
                                                      const float* __restrict__ qnw,
                                                      const float* __restrict__ knw,
                                                      const float* __restrict__ rope,
                                                      const int* __restrict__ pos_ids,
                                                      const int* __restrict__ start_idx,
                                                      float* __restrict__ kc,
                                                      float* __restrict__ vc) {
    int t = blockIdx.x, hh = blockIdx.y, dh = threadIdx.x;
    int past = start_idx[0];
    if (hh >= HQ_ + HKV_) {  // V: straight copy into cache slot
        int kh = hh - HQ_ - HKV_;
        vc[((size_t)kh * CACHE_ + past + t) * DH_ + dh] =
            vraw[(size_t)t * HKV_ * DH_ + kh * DH_ + dh];
        return;
    }
    __shared__ float buf[DH_];
    __shared__ float red[4];
    bool isq = hh < HQ_;
    const float* src = isq ? (qb + (size_t)t * HQ_ * DH_ + hh * DH_)
                           : (kraw + (size_t)t * HKV_ * DH_ + (hh - HQ_) * DH_);
    const float* nw = isq ? qnw : knw;
    float x = src[dh];
    float ss = x * x;
    for (int off = 16; off; off >>= 1) ss += __shfl_xor(ss, off);
    if ((threadIdx.x & 31) == 0) red[threadIdx.x >> 5] = ss;
    __syncthreads();
    float tot = red[0] + red[1] + red[2] + red[3];
    float xn = x * rsqrtf(tot / (float)DH_ + 1e-6f) * nw[dh];
    buf[dh] = xn;
    __syncthreads();
    int pos = pos_ids[t];
    float c = rope[(size_t)pos * DH_ + dh];
    float s = rope[(size_t)MAXPOS_ * DH_ + (size_t)pos * DH_ + dh];
    float rot = (dh < 64) ? -buf[dh + 64] : buf[dh - 64];
    float out = xn * c + rot * s;
    if (isq) {
        qb[(size_t)t * HQ_ * DH_ + hh * DH_ + dh] = out * 0.08838834764831845f;  // 1/sqrt(DH)
    } else {
        int kh = hh - HQ_;
        kc[((size_t)kh * CACHE_ + past + t) * DH_ + dh] = out;
    }
}

// ---------------- flash attention over 2112-entry cache (WMMA) ----------------
// grid (T/16, HQ), block 128 = 4 waves; each wave owns a strided set of 32-key chunks,
// online softmax in registers, P@V via WMMA with P staged through wave-private LDS.
// Score WMMAs alternate between two accumulators (one per 16-key tile) so consecutive
// WMMAs are independent -> no D->C RAW hazard NOPs.
__global__ __launch_bounds__(128) void attn_kernel(const float* __restrict__ qbuf,
                                                   const float* __restrict__ kc,
                                                   const float* __restrict__ vc,
                                                   float* __restrict__ obuf) {
    int qt = blockIdx.x, h = blockIdx.y;
    int kvh = h >> 1;  // GQA: 2 q-heads per kv-head
    const float* K = kc + (size_t)kvh * CACHE_ * DH_;
    const float* V = vc + (size_t)kvh * CACHE_ * DH_;
    int wave = threadIdx.x >> 5, lane = threadIdx.x & 31;
    int ln = lane & 15, hi = lane >> 4;

    __shared__ __bf16 Plds[4][16][34];
    __shared__ float Olds[4][16][DH_];
    __shared__ float Mlds[4][16], Llds[4][16];

    // q A-fragments (4 K-steps over DH=128); 1/sqrt(DH) already folded into qbuf
    v16bf aq[4];
    {
        int tok = qt * 16 + ln;
        const float* qp = qbuf + (size_t)tok * HQ_ * DH_ + h * DH_;
        int kb = hi ? 8 : 0;
        #pragma unroll
        for (int f = 0; f < 4; ++f)
            #pragma unroll
            for (int v = 0; v < 8; ++v) {
                int ko = a_koff(v, 0, kb);
                aq[f][2 * v + 0] = (__bf16)qp[f * 32 + ko];
                aq[f][2 * v + 1] = (__bf16)qp[f * 32 + ko + 1];
            }
    }
    float m[8], l[8];
    v8f oA[8];
    #pragma unroll
    for (int v = 0; v < 8; ++v) { m[v] = -1e30f; l[v] = 0.f; }
    #pragma unroll
    for (int nt = 0; nt < 8; ++nt) { v8f z = {}; oA[nt] = z; }

    const int nch = CACHE_ / 32;  // 66
    for (int c = wave; c < nch; c += 4) {
        int key0 = c * 32;
        int kbb = hi ? 16 : 0;
        // scores: two 16x16 tiles (keys key0..+15, key0+16..+31), interleaved accumulators
        v8f s0 = {}, s1 = {};
        {
            int keyA = key0 + ln;
            int keyB = key0 + 16 + ln;
            #pragma unroll
            for (int f = 0; f < 4; ++f) {
                v16bf bK0, bK1;
                #pragma unroll
                for (int v = 0; v < 8; ++v) {
                    int d0 = f * 32 + kbb + 2 * v;
                    bK0[2 * v + 0] = (__bf16)K[(size_t)keyA * DH_ + d0];
                    bK0[2 * v + 1] = (__bf16)K[(size_t)keyA * DH_ + d0 + 1];
                    bK1[2 * v + 0] = (__bf16)K[(size_t)keyB * DH_ + d0];
                    bK1[2 * v + 1] = (__bf16)K[(size_t)keyB * DH_ + d0 + 1];
                }
                s0 = __builtin_amdgcn_wmma_f32_16x16x32_bf16(
                    false, aq[f], false, bK0, (short)0, s0, false, false);
                s1 = __builtin_amdgcn_wmma_f32_16x16x32_bf16(
                    false, aq[f], false, bK1, (short)0, s1, false, false);
            }
        }
        // online softmax per row (row = v + 8*hi, keys striped across the 16-lane half)
        float p0[8], p1[8], scv[8];
        #pragma unroll
        for (int v = 0; v < 8; ++v) {
            float smax = fmaxf(s0[v], s1[v]);
            for (int off = 8; off; off >>= 1) smax = fmaxf(smax, __shfl_xor(smax, off));
            float mn = fmaxf(m[v], smax);
            float sc = __expf(m[v] - mn);
            float a0 = __expf(s0[v] - mn);
            float a1 = __expf(s1[v] - mn);
            float rs = a0 + a1;
            for (int off = 8; off; off >>= 1) rs += __shfl_xor(rs, off);
            l[v] = l[v] * sc + rs;
            m[v] = mn;
            p0[v] = a0; p1[v] = a1; scv[v] = sc;
        }
        #pragma unroll
        for (int nt = 0; nt < 8; ++nt)
            #pragma unroll
            for (int v = 0; v < 8; ++v) oA[nt][v] *= scv[v];
        // stage P (16q x 32key) into wave-private LDS in A-matrix-friendly order
        #pragma unroll
        for (int v = 0; v < 8; ++v) {
            int row = v + 8 * hi;
            Plds[wave][row][ln]      = (__bf16)p0[v];
            Plds[wave][row][16 + ln] = (__bf16)p1[v];
        }
        asm volatile("s_wait_dscnt 0" ::: "memory");  // intra-wave LDS RAW
        v16bf aP;
        {
            int kb = hi ? 8 : 0;
            #pragma unroll
            for (int v = 0; v < 8; ++v) {
                int ko = a_koff(v, 0, kb);
                aP[2 * v + 0] = Plds[wave][ln][ko];
                aP[2 * v + 1] = Plds[wave][ln][ko + 1];
            }
        }
        // O += P @ V_chunk (8 N-tiles over DH=128, independent accumulators)
        #pragma unroll
        for (int nt = 0; nt < 8; ++nt) {
            v16bf bV;
            int dcol = nt * 16 + ln;
            #pragma unroll
            for (int v = 0; v < 8; ++v) {
                size_t kr = (size_t)(key0 + kbb + 2 * v);
                bV[2 * v + 0] = (__bf16)V[kr * DH_ + dcol];
                bV[2 * v + 1] = (__bf16)V[(kr + 1) * DH_ + dcol];
            }
            oA[nt] = __builtin_amdgcn_wmma_f32_16x16x32_bf16(
                false, aP, false, bV, (short)0, oA[nt], false, false);
        }
    }
    // publish per-wave partials, then merge across the 4 waves
    #pragma unroll
    for (int v = 0; v < 8; ++v) {
        int row = v + 8 * hi;
        if (ln == 0) { Mlds[wave][row] = m[v]; Llds[wave][row] = l[v]; }
        #pragma unroll
        for (int nt = 0; nt < 8; ++nt) Olds[wave][row][nt * 16 + ln] = oA[nt][v];
    }
    __syncthreads();
    for (int i = threadIdx.x; i < 16 * DH_; i += 128) {
        int row = i >> 7, dh = i & 127;
        float M = fmaxf(fmaxf(Mlds[0][row], Mlds[1][row]),
                        fmaxf(Mlds[2][row], Mlds[3][row]));
        float Lt = 0.f, acc = 0.f;
        #pragma unroll
        for (int w = 0; w < 4; ++w) {
            float e = __expf(Mlds[w][row] - M);
            Lt += Llds[w][row] * e;
            acc += Olds[w][row][dh] * e;
        }
        int tok = qt * 16 + row;
        obuf[(size_t)tok * HQ_ * DH_ + h * DH_ + dh] = acc / Lt;
    }
}

// g = silu(g) * u
__global__ void silu_mul_kernel(float* __restrict__ g, const float* __restrict__ u, int n) {
    int i = blockIdx.x * 256 + threadIdx.x;
    if (i < n) {
        float x = g[i];
        g[i] = (x / (1.f + __expf(-x))) * u[i];
    }
}

// final norm + W_out + Euler step
__global__ __launch_bounds__(256) void final_out_kernel(const float* __restrict__ h,
                                                        const float* __restrict__ fnw,
                                                        const float* __restrict__ Wout,
                                                        const float* __restrict__ bout,
                                                        const float* __restrict__ noise,
                                                        const float* __restrict__ t0,
                                                        const float* __restrict__ t1,
                                                        float* __restrict__ den) {
    int t = blockIdx.x, tid = threadIdx.x;
    __shared__ float red[8];
    __shared__ float pr[8][8];
    float ss = 0.f;
    for (int d = tid; d < D_; d += 256) { float v = h[(size_t)t * D_ + d]; ss += v * v; }
    for (int off = 16; off; off >>= 1) ss += __shfl_xor(ss, off);
    if ((tid & 31) == 0) red[tid >> 5] = ss;
    __syncthreads();
    float tot = 0.f;
    #pragma unroll
    for (int w = 0; w < 8; ++w) tot += red[w];
    float inv = rsqrtf(tot / (float)D_ + 1e-6f);
    float pa[A_];
    #pragma unroll
    for (int a = 0; a < A_; ++a) pa[a] = 0.f;
    for (int d = tid; d < D_; d += 256) {
        float xn = h[(size_t)t * D_ + d] * inv * fnw[d];
        #pragma unroll
        for (int a = 0; a < A_; ++a) pa[a] += xn * Wout[(size_t)d * A_ + a];
    }
    #pragma unroll
    for (int a = 0; a < A_; ++a)
        for (int off = 16; off; off >>= 1) pa[a] += __shfl_xor(pa[a], off);
    if ((tid & 31) == 0)
        #pragma unroll
        for (int a = 0; a < A_; ++a) pr[tid >> 5][a] = pa[a];
    __syncthreads();
    if (tid < A_) {
        float s = bout[tid];
        #pragma unroll
        for (int w = 0; w < 8; ++w) s += pr[w][tid];
        float dt = t1[0] - t0[0];
        den[t * A_ + tid] = noise[t * A_ + tid] + dt * s;
    }
}

// ---------------- host orchestration ----------------
extern "C" void kernel_launch(void* const* d_in, const int* in_sizes, int n_in,
                              void* d_out, int out_size, void* d_ws, size_t ws_size,
                              hipStream_t stream) {
    (void)in_sizes; (void)n_in; (void)out_size; (void)ws_size;
    const float* noise   = (const float*)d_in[0];
    const float* t0      = (const float*)d_in[1];
    const float* t1      = (const float*)d_in[2];
    const float* rope    = (const float*)d_in[3];
    const int*   pos_ids = (const int*)d_in[4];
    const float* kci     = (const float*)d_in[5];
    const float* vci     = (const float*)d_in[6];
    const int*   start   = (const int*)d_in[7];
    const float* W_in    = (const float*)d_in[8];
    const float* b_in    = (const float*)d_in[9];
    const float* Wt1     = (const float*)d_in[10];
    const float* Wt2     = (const float*)d_in[11];
    const float* ln1     = (const float*)d_in[12];
    const float* qnw     = (const float*)d_in[13];
    const float* knw     = (const float*)d_in[14];
    const float* Wq      = (const float*)d_in[15];
    const float* Wk      = (const float*)d_in[16];
    const float* Wv      = (const float*)d_in[17];
    const float* Wo      = (const float*)d_in[18];
    const float* ln2     = (const float*)d_in[19];
    const float* Wg      = (const float*)d_in[20];
    const float* Wu      = (const float*)d_in[21];
    const float* Wd      = (const float*)d_in[22];
    const float* fnw     = (const float*)d_in[23];
    const float* Wout    = (const float*)d_in[24];
    const float* bout    = (const float*)d_in[25];

    const size_t cacheElems = (size_t)L_ * HKV_ * CACHE_ * DH_;  // per cache tensor
    float* outp = (float*)d_out;
    float* den  = outp;                 // [T, A]
    float* pk   = outp + T_ * A_;       // [L,1,HKV,CACHE,DH]
    float* pv   = pk + cacheElems;

    float* ws   = (float*)d_ws;
    float* h    = ws;                         // [64, 2048]
    float* hn   = h    + 64 * 2048;           // [64, 2048]
    float* qb   = hn   + 64 * 2048;           // [64, 16*128]
    float* kraw = qb   + 64 * 2048;           // [64, 8*128]
    float* vraw = kraw + 64 * 1024;           // [64, 8*128]
    float* ob   = vraw + 64 * 1024;           // [64, 16*128]
    float* gate = ob   + 64 * 2048;           // [64, 6144]
    float* up   = gate + 64 * 6144;           // [64, 6144]
    float* temb = up   + 64 * 6144;           // [2048]
    float* th   = temb + 2048;                // [2048]

    // 1) seed output caches with the input caches (updated rows overwritten below)
    int n4 = (int)(cacheElems / 4);
    copy4_kernel<<<(n4 + 255) / 256, 256, 0, stream>>>((const float4*)kci, (float4*)pk, n4);
    copy4_kernel<<<(n4 + 255) / 256, 256, 0, stream>>>((const float4*)vci, (float4*)pv, n4);

    // 2) time embedding + input projection
    time_mlp1_kernel<<<D_ / 256, 256, 0, stream>>>(t0, Wt1, th);
    time_mlp2_kernel<<<D_ / 256, 256, 0, stream>>>(th, Wt2, temb);
    h_init_kernel<<<dim3(D_ / 256, T_), 256, 0, stream>>>(noise, W_in, b_in, temb, h);

    for (int l = 0; l < L_; ++l) {
        const float* Wq_l = Wq + (size_t)l * D_ * (HQ_ * DH_);
        const float* Wk_l = Wk + (size_t)l * D_ * (HKV_ * DH_);
        const float* Wv_l = Wv + (size_t)l * D_ * (HKV_ * DH_);
        const float* Wo_l = Wo + (size_t)l * (HQ_ * DH_) * D_;
        const float* Wg_l = Wg + (size_t)l * D_ * F_;
        const float* Wu_l = Wu + (size_t)l * D_ * F_;
        const float* Wd_l = Wd + (size_t)l * F_ * D_;
        float* pk_l = pk + (size_t)l * HKV_ * CACHE_ * DH_;
        float* pv_l = pv + (size_t)l * HKV_ * CACHE_ * DH_;

        rmsnorm_kernel<<<T_, 256, 0, stream>>>(h, ln1 + (size_t)l * D_, hn);
        gemm64_wmma<false><<<(HQ_ * DH_) / 128, 256, 0, stream>>>(hn, Wq_l, qb, D_, HQ_ * DH_);
        gemm64_wmma<false><<<(HKV_ * DH_) / 128, 256, 0, stream>>>(hn, Wk_l, kraw, D_, HKV_ * DH_);
        gemm64_wmma<false><<<(HKV_ * DH_) / 128, 256, 0, stream>>>(hn, Wv_l, vraw, D_, HKV_ * DH_);
        qk_post_kernel<<<dim3(T_, HQ_ + 2 * HKV_), 128, 0, stream>>>(
            qb, kraw, vraw, qnw + (size_t)l * DH_, knw + (size_t)l * DH_,
            rope, pos_ids, start, pk_l, pv_l);
        attn_kernel<<<dim3(T_ / 16, HQ_), 128, 0, stream>>>(qb, pk_l, pv_l, ob);
        gemm64_wmma<true><<<D_ / 128, 256, 0, stream>>>(ob, Wo_l, h, HQ_ * DH_, D_);

        rmsnorm_kernel<<<T_, 256, 0, stream>>>(h, ln2 + (size_t)l * D_, hn);
        gemm64_wmma<false><<<F_ / 128, 256, 0, stream>>>(hn, Wg_l, gate, D_, F_);
        gemm64_wmma<false><<<F_ / 128, 256, 0, stream>>>(hn, Wu_l, up, D_, F_);
        silu_mul_kernel<<<(T_ * F_) / 256, 256, 0, stream>>>(gate, up, T_ * F_);
        gemm64_wmma<true><<<D_ / 128, 256, 0, stream>>>(gate, Wd_l, h, F_, D_);
    }

    final_out_kernel<<<T_, 256, 0, stream>>>(h, fnw, Wout, bout, noise, t0, t1, den);
}
